// keras_multihead_attentive_layer_58222576664642
// MI455X (gfx1250) — matrix operand order repaired and checked by reading, not compiled, not run on hardware
//
#include <hip/hip_runtime.h>
#include <hip/hip_bf16.h>
#include <math.h>

// Problem constants (keras_multihead_attentive_layer)
#define B_ 8
#define S_ 2048
#define D_ 512
#define H_ 8
#define E_ 64
#define O_ 512
#define F_ (H_ * E_)   // 512, concat-head feature dim

typedef __attribute__((ext_vector_type(16))) _Float16 v16h;
typedef __attribute__((ext_vector_type(8)))  _Float16 v8h;
typedef __attribute__((ext_vector_type(8)))  float    v8f;
typedef int v4i __attribute__((vector_size(16)));

#define AS1 __attribute__((address_space(1)))
#define AS3 __attribute__((address_space(3)))

// --------------------------- async-to-LDS support ---------------------------
#if defined(__has_builtin)
#  if __has_builtin(__builtin_amdgcn_global_load_async_to_lds_b128)
#    define USE_ASYNC_LDS 1
#  else
#    define USE_ASYNC_LDS 0
#  endif
#else
#  define USE_ASYNC_LDS 0
#endif

#if USE_ASYNC_LDS
#  if __has_builtin(__builtin_amdgcn_s_wait_asynccnt)
#    define WAIT_ASYNC(n) __builtin_amdgcn_s_wait_asynccnt(n)
#  else
#    define WAIT_ASYNC(n) asm volatile("s_wait_asynccnt " #n ::: "memory")
#  endif
#else
#  define WAIT_ASYNC(n) ((void)0)
#endif

// Copy 8 halfs (16B) global -> LDS.  Async path uses the CDNA5 async DMA
// (tracked by ASYNCcnt); fallback is a synchronous b128 register bounce.
static __device__ __forceinline__ void copy16_g2l(const _Float16* g, _Float16* l) {
#if USE_ASYNC_LDS
  // prototype (from clang diagnostic): (v4i AS1*, v4i AS3*, imm offset, imm cpol)
  __builtin_amdgcn_global_load_async_to_lds_b128(
      (AS1 v4i*)(unsigned long long)g,   // global flat addr == generic numerically
      (AS3 v4i*)l,                       // addrspacecast generic -> LDS
      0, 0);
#else
  *(v8h*)l = *(const v8h*)g;
#endif
}

// ---------------------------------------------------------------------------
// D = A(16x32,f16) * B(32x16,f16) + C(16x16,f32)   (CDNA5 wave32 WMMA)
// ---------------------------------------------------------------------------
static __device__ __forceinline__ v8f wmma16(v16h a, v16h b, v8f c) {
  return __builtin_amdgcn_wmma_f32_16x16x32_f16(
      /*neg_a=*/false, a, /*neg_b=*/false, b,
      /*c_mod=*/(short)0, c, /*reuse_a=*/false, /*reuse_b=*/false);
}

// Two 8-half (16B) loads at p and p+16; p must be 16B aligned.
// Used for B fragments: elements 0..7 -> K = kbB + i, 8..15 -> K = kbB + 16 + (i-8) (kbB = 0 lanes 0-15, 8 lanes 16-31).
static __device__ __forceinline__ v16h ld_contig16(const _Float16* p) {
  const v8h* q = (const v8h*)p;
  v8h lo = q[0];
  v8h hi = q[2];
  v16h r;
#pragma unroll
  for (int i = 0; i < 8; ++i) { r[i] = lo[i]; r[8 + i] = hi[i]; }
  return r;
}

// A fragment (16x32 f16, ISA 7.12.2): rowp = start of this lane's row (m = lane&15);
// kbase = chunk_base + (lane<16 ? 0 : 8).
// elements 0..7 -> K = kbase..kbase+7 ; elements 8..15 -> K = kbase+16..kbase+23
static __device__ __forceinline__ v16h ld_a_frag(const _Float16* rowp, int kbase) {
  const v8h* q0 = (const v8h*)(rowp + kbase);
  const v8h* q1 = (const v8h*)(rowp + kbase + 16);
  v8h lo = *q0;
  v8h hi = *q1;
  v16h r;
#pragma unroll
  for (int i = 0; i < 8; ++i) { r[i] = lo[i]; r[8 + i] = hi[i]; }
  return r;
}

// ---------------------------------------------------------------------------
// Kernel 0: fp32 -> f16 weight conversion with transpose for contiguous
// B-fragment loads later.  Wq/Wk/Wv [H,D,E] -> [H,E,D];  Wo [F,O] -> [O,F].
// ---------------------------------------------------------------------------
__global__ void cvt_weights(const float* __restrict__ Wq, const float* __restrict__ Wk,
                            const float* __restrict__ Wv, const float* __restrict__ Wo,
                            _Float16* __restrict__ Wqt, _Float16* __restrict__ Wkt,
                            _Float16* __restrict__ Wvt, _Float16* __restrict__ Wot) {
  int i = blockIdx.x * blockDim.x + threadIdx.x;
  if (i >= H_ * D_ * E_) return;
  int h   = i / (D_ * E_);
  int rem = i - h * (D_ * E_);
  int d   = rem / E_;
  int e   = rem - d * E_;
  int t   = h * (E_ * D_) + e * D_ + d;
  Wqt[t] = (_Float16)Wq[i];
  Wkt[t] = (_Float16)Wk[i];
  Wvt[t] = (_Float16)Wv[i];
  // Wo: same element count (F_*O_ == H_*D_*E_ == 262144)
  int f = i / O_;
  int o = i - f * O_;
  Wot[o * F_ + f] = (_Float16)Wo[i];
}

// ---------------------------------------------------------------------------
// Kernel 1: QKV projection.  One wave computes a 16x64 (rows x E) tile for one
// head / one matrix.  A = X tile (fp32 converted in-register), B = W^T chunk.
// Q,K stored [B,H,S,E]; V stored transposed [B,H,E,S] (contiguous PV B-frags).
// ---------------------------------------------------------------------------
__global__ __launch_bounds__(128) void qkv_proj(
    const float* __restrict__ X,
    const _Float16* __restrict__ Wqt, const _Float16* __restrict__ Wkt,
    const _Float16* __restrict__ Wvt,
    _Float16* __restrict__ Qh, _Float16* __restrict__ Kh,
    _Float16* __restrict__ Vt) {
  const int lane  = threadIdx.x & 31;
  const int wave  = threadIdx.x >> 5;
  const int blk   = blockIdx.x * 4 + wave;   // row tile over [B*S): 0..1023
  const int h     = blockIdx.y;
  const int which = blockIdx.z;              // 0=Q 1=K 2=V

  const int nn    = lane & 15;
  const int mbase = (lane < 16) ? 0 : 8;
  const int kbA   = (lane < 16) ? 0 : 8;
  const int kbB   = (lane < 16) ? 0 : 8;

  const int row0 = blk * 16;
  const int b    = row0 / S_;
  const int sb   = row0 % S_;

  const _Float16* W  = (which == 0) ? Wqt : (which == 1) ? Wkt : Wvt;
  const _Float16* Wh = W + (size_t)h * (E_ * D_);       // [E,D]
  const float*  xrow = X + (size_t)(row0 + nn) * D_;

  v8f acc[4] = {};
#pragma unroll 2
  for (int k0 = 0; k0 < D_; k0 += 32) {
    v16h a;
    {
      const float* p0 = xrow + k0 + kbA;
      const float* p1 = xrow + k0 + kbA + 16;
#pragma unroll
      for (int i = 0; i < 8; ++i) { a[i] = (_Float16)p0[i]; a[8 + i] = (_Float16)p1[i]; }
    }
#pragma unroll
    for (int nt = 0; nt < 4; ++nt) {
      v16h bb = ld_contig16(Wh + (size_t)(nt * 16 + nn) * D_ + k0 + kbB);
      acc[nt] = wmma16(a, bb, acc[nt]);
    }
  }

  if (which == 2) {
    // V transposed: Vt[b][h][e][s]; each lane's 8 rows are contiguous in s.
    _Float16* vb = Vt + (size_t)(b * H_ + h) * E_ * S_;
#pragma unroll
    for (int nt = 0; nt < 4; ++nt) {
      v8h t;
#pragma unroll
      for (int r = 0; r < 8; ++r) t[r] = (_Float16)acc[nt][r];
      *(v8h*)(vb + (size_t)(nt * 16 + nn) * S_ + sb + mbase) = t;
    }
  } else {
    _Float16* dst = ((which == 0) ? Qh : Kh) + ((size_t)(b * H_ + h) * S_ + sb) * E_;
#pragma unroll
    for (int nt = 0; nt < 4; ++nt)
#pragma unroll
      for (int r = 0; r < 8; ++r)
        dst[(size_t)(mbase + r) * E_ + nt * 16 + nn] = (_Float16)acc[nt][r];
  }
}

// ---------------------------------------------------------------------------
// Kernel 2: causal flash attention.
// Block = 4 waves on one (b,h), query blocks s0 = (4*blockIdx.x + wave)*16.
// The 4 waves SHARE 32-key K/V tiles, double-buffered in LDS and filled with
// CDNA5 async global->LDS DMA (ASYNCcnt), cutting L2 traffic 4x.
// Per 32 keys per wave: 4 WMMA scores + online softmax (cross-lane reductions
// in 16-lane halves, matching the C-matrix layout) + P restaged via LDS
// (C-layout -> A-layout) + 4 WMMA for P*V.
// ---------------------------------------------------------------------------
__global__ __launch_bounds__(128) void attn(
    const _Float16* __restrict__ Qh, const _Float16* __restrict__ Kh,
    const _Float16* __restrict__ Vt, _Float16* __restrict__ Ctx) {
  __shared__ __align__(16) _Float16 ldsK[2][32 * E_];   // keys: [t][e], 2 x 4KB
  __shared__ __align__(16) _Float16 ldsV[2][E_ * 32];   // values: [e][t], 2 x 4KB
  __shared__ __align__(16) _Float16 ldsP[4][16 * 32];   // per-wave P staging

  const int tid   = threadIdx.x;
  const int lane  = tid & 31;
  const int wave  = tid >> 5;
  const int nn    = lane & 15;
  const int mbase = (lane < 16) ? 0 : 8;
  const int kbA   = (lane < 16) ? 0 : 8;
  const int kbB   = (lane < 16) ? 0 : 8;

  const int bh   = blockIdx.y;               // b*H + h
  const int qg   = blockIdx.x;               // query group (64 rows)
  const int s0   = (qg * 4 + wave) * 16;     // this wave's query block
  const int tmax = (qg + 1) * 64;            // keys needed by the whole block
  const int bidx = bh / H_;
  const int h    = bh % H_;

  const _Float16* Qb = Qh + (size_t)bh * S_ * E_;
  const _Float16* Kb = Kh + (size_t)bh * S_ * E_;
  const _Float16* Vb = Vt + (size_t)bh * E_ * S_;

  // Q A-fragments for the two E-chunks (loaded once per block)
  const _Float16* qrow = Qb + (size_t)(s0 + nn) * E_;
  v16h qa0 = ld_a_frag(qrow, 0 + kbA);
  v16h qa1 = ld_a_frag(qrow, 32 + kbA);

  float mrow[8], lrow[8];
  v8f o[4] = {};
#pragma unroll
  for (int r = 0; r < 8; ++r) { mrow[r] = -INFINITY; lrow[r] = 0.f; }

  _Float16* pt = ldsP[wave];

  // Cooperative 32-key K/V tile fill: 4 x b128 per thread per batch.
  //   K tile: 32 rows x 128B  -> 256 chunks,  V tile: 64 rows x 64B -> 256 chunks.
  auto load_kv = [&](int t0, _Float16* lk, _Float16* lv) {
#pragma unroll
    for (int p = 0; p < 2; ++p) {
      int l  = tid + p * 128;
      int j  = l >> 3, c  = l & 7;    // K: key row j, 16B chunk c
      copy16_g2l(Kb + (size_t)(t0 + j) * E_ + c * 8, lk + j * E_ + c * 8);
      int e  = l >> 2, c2 = l & 3;    // V: e row, 16B chunk c2
      copy16_g2l(Vb + (size_t)e * S_ + t0 + c2 * 8, lv + e * 32 + c2 * 8);
    }
  };

  // Prologue: batches 0 and 1 (tmax >= 64, so both exist).
  load_kv(0, ldsK[0], ldsV[0]);
  load_kv(32, ldsK[1], ldsV[1]);

  for (int t0 = 0, it = 0; t0 < tmax; t0 += 32, ++it) {
    const int cur = it & 1;
    // Async completions are in-order: <=4 outstanding means batch `it` landed.
    if (t0 + 32 < tmax) { WAIT_ASYNC(4); } else { WAIT_ASYNC(0); }
    __syncthreads();                               // tile `cur` visible to all waves

    if (t0 < s0 + 16) {                            // wave-uniform causal extent
      const _Float16* kt = ldsK[cur];
      const _Float16* vt = ldsV[cur];

      // --- scores: two 16x16 tiles (keys t0..t0+15, t0+16..t0+31) ---
      v8f sc[2] = {};
#pragma unroll
      for (int hlf = 0; hlf < 2; ++hlf) {
        const _Float16* krow = kt + (size_t)(hlf * 16 + nn) * E_;
        v16h bk0 = ld_contig16(krow + 0 + kbB);
        v16h bk1 = ld_contig16(krow + 32 + kbB);
        sc[hlf] = wmma16(qa0, bk0, sc[hlf]);
        sc[hlf] = wmma16(qa1, bk1, sc[hlf]);
      }

      // --- causal mask (only diagonal-region iterations) ---
      if (t0 + 31 > s0) {
#pragma unroll
        for (int hlf = 0; hlf < 2; ++hlf) {
          int t = t0 + hlf * 16 + nn;
#pragma unroll
          for (int r = 0; r < 8; ++r) {
            int s = s0 + mbase + r;
            if (t > s) sc[hlf][r] = -1e9f;
          }
        }
      }

      // --- online softmax (row = 16 lanes of one half-wave, per C-layout) ---
      float pl[2][8];
      float corr[8];
#pragma unroll
      for (int r = 0; r < 8; ++r) {
        float cmax = fmaxf(sc[0][r], sc[1][r]);
        cmax = fmaxf(cmax, __shfl_xor(cmax, 1, 32));
        cmax = fmaxf(cmax, __shfl_xor(cmax, 2, 32));
        cmax = fmaxf(cmax, __shfl_xor(cmax, 4, 32));
        cmax = fmaxf(cmax, __shfl_xor(cmax, 8, 32));
        float mnew = fmaxf(mrow[r], cmax);
        corr[r] = __expf(mrow[r] - mnew);
        mrow[r] = mnew;
        float p0 = __expf(sc[0][r] - mnew);
        float p1 = __expf(sc[1][r] - mnew);
        pl[0][r] = p0;
        pl[1][r] = p1;
        float ssum = p0 + p1;
        ssum += __shfl_xor(ssum, 1, 32);
        ssum += __shfl_xor(ssum, 2, 32);
        ssum += __shfl_xor(ssum, 4, 32);
        ssum += __shfl_xor(ssum, 8, 32);
        lrow[r] = lrow[r] * corr[r] + ssum;
      }
#pragma unroll
      for (int nt = 0; nt < 4; ++nt)
#pragma unroll
        for (int r = 0; r < 8; ++r) o[nt][r] *= corr[r];

      // --- stage P (16x32 f16) in LDS: C-layout -> A-layout transpose ---
#pragma unroll
      for (int hlf = 0; hlf < 2; ++hlf)
#pragma unroll
        for (int r = 0; r < 8; ++r)
          pt[(mbase + r) * 32 + hlf * 16 + nn] = (_Float16)pl[hlf][r];
      __asm__ volatile("s_wait_dscnt 0" ::: "memory");   // same-wave DS RAW

      v16h pa = ld_a_frag(pt + nn * 32, kbA);

      // --- PV: ctx += P(16x32) * V(32xE) ---
#pragma unroll
      for (int nt = 0; nt < 4; ++nt) {
        v16h bv = ld_contig16(vt + (size_t)(nt * 16 + nn) * 32 + kbB);
        o[nt] = wmma16(pa, bv, o[nt]);
      }
    }

    __syncthreads();                               // all waves done reading `cur`
    if (t0 + 64 < tmax)
      load_kv(t0 + 64, ldsK[cur], ldsV[cur]);      // refill `cur` (batch it+2)
  }

  // --- normalize, store ctx as [B, S, H*E] f16 (head-major concat) ---
  _Float16* cb = Ctx + ((size_t)bidx * S_ + s0) * F_ + h * E_;
#pragma unroll
  for (int nt = 0; nt < 4; ++nt)
#pragma unroll
    for (int r = 0; r < 8; ++r) {
      float v = o[nt][r] / lrow[r];
      cb[(size_t)(mbase + r) * F_ + nt * 16 + nn] = (_Float16)v;
    }
}

// ---------------------------------------------------------------------------
// Kernel 3: output projection  out[B*S, O] = ctx[B*S, F] * Wo[F, O], fp32 out.
// One wave = 16x64 output tile.
// ---------------------------------------------------------------------------
__global__ __launch_bounds__(128) void out_proj(
    const _Float16* __restrict__ Ctx, const _Float16* __restrict__ Wot,
    float* __restrict__ Out) {
  const int lane  = threadIdx.x & 31;
  const int wave  = threadIdx.x >> 5;
  const int lin   = blockIdx.x * 4 + wave;   // 0..8191
  const int rt    = lin >> 3;                // row tile 0..1023
  const int ct    = lin & 7;                 // 64-wide col tile 0..7
  const int nn    = lane & 15;
  const int mbase = (lane < 16) ? 0 : 8;
  const int kbA   = (lane < 16) ? 0 : 8;
  const int kbB   = (lane < 16) ? 0 : 8;

  const int row0 = rt * 16;
  const int o0   = ct * 64;
  const _Float16* crow = Ctx + (size_t)(row0 + nn) * F_;

  v8f acc[4] = {};
#pragma unroll 2
  for (int k0 = 0; k0 < F_; k0 += 32) {
    v16h a = ld_a_frag(crow, k0 + kbA);
#pragma unroll
    for (int nt = 0; nt < 4; ++nt) {
      v16h bb = ld_contig16(Wot + (size_t)(o0 + nt * 16 + nn) * F_ + k0 + kbB);
      acc[nt] = wmma16(a, bb, acc[nt]);
    }
  }

  float* orow = Out + (size_t)row0 * O_ + o0;
#pragma unroll
  for (int nt = 0; nt < 4; ++nt)
#pragma unroll
    for (int r = 0; r < 8; ++r)
      orow[(size_t)(mbase + r) * O_ + nt * 16 + nn] = acc[nt][r];
}

// ---------------------------------------------------------------------------
extern "C" void kernel_launch(void* const* d_in, const int* in_sizes, int n_in,
                              void* d_out, int out_size, void* d_ws, size_t ws_size,
                              hipStream_t stream) {
  (void)in_sizes; (void)n_in; (void)out_size; (void)ws_size;
  const float* query = (const float*)d_in[0];
  const float* Wq    = (const float*)d_in[1];
  const float* Wk    = (const float*)d_in[2];
  const float* Wv    = (const float*)d_in[3];
  const float* Wo    = (const float*)d_in[4];
  float*       out   = (float*)d_out;

  // f16 workspace layout (~69 MB total)
  _Float16* ws = (_Float16*)d_ws;
  const size_t WSZ = (size_t)H_ * D_ * E_;        // 262144 halfs per weight
  const size_t TSZ = (size_t)B_ * H_ * S_ * E_;   // 8388608 halfs per tensor
  _Float16* Wqt = ws;
  _Float16* Wkt = Wqt + WSZ;
  _Float16* Wvt = Wkt + WSZ;
  _Float16* Wot = Wvt + WSZ;
  _Float16* Qh  = Wot + WSZ;
  _Float16* Kh  = Qh + TSZ;
  _Float16* Vt  = Kh + TSZ;     // [B,H,E,S]
  _Float16* Cx  = Vt + TSZ;     // [B,S,H*E]

  cvt_weights<<<dim3((H_ * D_ * E_ + 255) / 256), dim3(256), 0, stream>>>(
      Wq, Wk, Wv, Wo, Wqt, Wkt, Wvt, Wot);

  // 1024 row-tiles / 4 waves per block
  qkv_proj<<<dim3(256, H_, 3), dim3(128), 0, stream>>>(
      query, Wqt, Wkt, Wvt, Qh, Kh, Vt);

  // 32 query groups (4 blocks of 16 rows each) per (b,h)
  attn<<<dim3(32, B_ * H_), dim3(128), 0, stream>>>(Qh, Kh, Vt, Cx);

  // 1024 row-tiles x 8 col-tiles = 8192 waves / 4
  out_proj<<<dim3(2048), dim3(128), 0, stream>>>(Cx, Wot, out);
}
